// Fine_Net_30099130810426
// MI455X (gfx1250) — matrix operand order repaired and checked by reading, not compiled, not run on hardware
//
#include <hip/hip_runtime.h>
#include <cmath>

#define NKP  2048
#define WINW 5
#define LW   25            // window points per keypoint
#define NTOK (NKP * LW)    // 51200 rows in flattened GEMMs
#define FD   128
#define HD   512           // up-sampled desc map spatial size
#define IMGW 1024
#define ATT_EPS 1e-6f

typedef _Float16 f16;
typedef __attribute__((ext_vector_type(16))) _Float16 v16h;
typedef __attribute__((ext_vector_type(8)))  float    v8f;

// ---------------------------------------------------------------- utilities
__global__ void cvt_f16_kernel(const float* __restrict__ s, f16* __restrict__ d, int n) {
    int i = blockIdx.x * 256 + threadIdx.x;
    if (i < n) d[i] = (f16)s[i];
}

__device__ __forceinline__ float elu1(float x) {
    return x > 0.f ? x + 1.f : __expf(x);   // elu(x)+1
}

// CDNA5 async memory->LDS copy (tracked by ASYNCcnt), 16B per lane.
// dst: LDS byte address (low 32 bits of generic pointer), src: 64-bit global addr.
__device__ __forceinline__ void async_ld_b128(unsigned lds_addr, const void* gaddr) {
    asm volatile("global_load_async_to_lds_b128 %0, %1, off"
                 :: "v"(lds_addr), "v"(gaddr) : "memory");
}
#define WAIT_ASYNC(n) asm volatile("s_wait_asynccnt %0" :: "n"(n) : "memory")

__device__ __forceinline__ unsigned lds_addr_of(const void* p) {
    return (unsigned)(unsigned long long)p;
}

// ------------------------------------------------- window gen + bilinear sample
// one block (128 threads = channels) per sample point
__global__ __launch_bounds__(128)
void sample_kernel(const float* __restrict__ kpts, const float* __restrict__ dmap,
                   float* __restrict__ dout, float* __restrict__ mask,
                   float* __restrict__ wins) {
    int p = blockIdx.x;            // 0 .. NTOK-1
    int c = threadIdx.x;           // channel
    int kp = p / LW, wi = p % LW;
    float wx = kpts[kp * 2 + 0] + (float)(wi % WINW - 2) * 2.0f;
    float wy = kpts[kp * 2 + 1] + (float)(wi / WINW - 2) * 2.0f;
    if (c == 0) {
        wins[(size_t)p * 2 + 0] = wx;
        wins[(size_t)p * 2 + 1] = wy;
        mask[p] = (wx >= 0.f && wx <= (float)(IMGW - 1) &&
                   wy >= 0.f && wy <= (float)(IMGW - 1)) ? 1.f : 0.f;
    }
    // SuperPoint normalization, s=2: g = (w - 0.5)/1022.5 * (HD-1)
    const float sc = (float)(HD - 1) / 1022.5f;
    float gx = (wx - 0.5f) * sc, gy = (wy - 0.5f) * sc;
    float fx0 = floorf(gx), fy0 = floorf(gy);
    float fx = gx - fx0, fy = gy - fy0;
    int x0 = (int)fx0, y0 = (int)fy0;
    const float* plane = dmap + (size_t)c * HD * HD;
    float acc = 0.f;
#pragma unroll
    for (int dy = 0; dy < 2; ++dy)
#pragma unroll
        for (int dx = 0; dx < 2; ++dx) {
            int xi = x0 + dx, yi = y0 + dy;
            float w = (dx ? fx : 1.f - fx) * (dy ? fy : 1.f - fy);
            bool valid = (xi >= 0 && xi < HD && yi >= 0 && yi < HD);
            int cx = min(max(xi, 0), HD - 1), cy = min(max(yi, 0), HD - 1);
            acc += valid ? w * plane[(size_t)cy * HD + cx] : 0.f;
        }
    dout[(size_t)p * FD + c] = acc;
}

// ---------------------------------------------------------------- WMMA GEMM
// C[M,N] = act( A[M,K] @ W[N,K]^T ).  M%128==0, N%64==0, K in {128,256} (template).
// 128 threads = 4 waves; block tile 128x64; wave tile 32x64 (2x4 wmma accumulators).
// Tiles staged with async memory->LDS copies, double-buffered; per wave one stage
// is exactly 10 async-load instructions (8 A + 2 B); async loads complete in
// order, so s_wait_asynccnt 10 retires exactly the previous stage.
#define LDSA_STRIDE 36   // f32, 144B row = 9 x 16B
#define LDSB_STRIDE 40   // f16,  80B row = 5 x 16B
template <int K, bool RELU>
__global__ __launch_bounds__(128)
void gemm_wmma_kernel(const float* __restrict__ A, const f16* __restrict__ W,
                      float* __restrict__ C, int N) {
    __shared__ __align__(16) float As[2][128][LDSA_STRIDE];
    __shared__ __align__(16) f16   Bs[2][64][LDSB_STRIDE];
    constexpr unsigned ASZ = 128 * LDSA_STRIDE * 4;   // bytes per A buffer
    constexpr unsigned BSZ = 64 * LDSB_STRIDE * 2;    // bytes per B buffer
    const int tid = threadIdx.x;
    const int wave = tid >> 5;
    const int lane = tid & 31;
    const int rr = lane & 15, hi = lane >> 4;
    const int mBase = blockIdx.y * 128;
    const int nBase = blockIdx.x * 64;

    // precompute per-thread staging addresses (branch-free stages in the K loop)
    const float* gA[8]; unsigned lA[8];
#pragma unroll
    for (int it = 0; it < 8; ++it) {              // A: 128 rows x 32 f32 = 1024 b128 chunks
        int i = tid + it * 128, r = i >> 3, ck = i & 7;
        gA[it] = A + (size_t)(mBase + r) * K + ck * 4;
        lA[it] = lds_addr_of(&As[0][r][ck * 4]);
    }
    const f16* gB[2]; unsigned lB[2];
#pragma unroll
    for (int it = 0; it < 2; ++it) {              // B: 64 rows x 32 f16 = 256 b128 chunks
        int i = tid + it * 128, r = i >> 2, ck = i & 3;
        gB[it] = W + (size_t)(nBase + r) * K + ck * 8;
        lB[it] = lds_addr_of(&Bs[0][r][ck * 8]);
    }
    auto stage = [&](int buf, int k0) {
#pragma unroll
        for (int it = 0; it < 8; ++it) async_ld_b128(lA[it] + buf * ASZ, gA[it] + k0);
#pragma unroll
        for (int it = 0; it < 2; ++it) async_ld_b128(lB[it] + buf * BSZ, gB[it] + k0);
    };

    v8f acc[2][4] = {};
    stage(0, 0);
#pragma unroll
    for (int ks = 0; ks < K / 32; ++ks) {
        const int cur = ks & 1;
        if (ks + 1 < K / 32) {
            stage(cur ^ 1, (ks + 1) * 32);   // prefetch next slab into other buffer
            WAIT_ASYNC(10);                  // retire previous stage only
        } else {
            WAIT_ASYNC(0);
        }
        __syncthreads();
        // A fragments: row M = lane&15; lanes 0-15 K={0..7,16..23}, 16-31 K={8..15,24..31}
        v16h a[2];
#pragma unroll
        for (int f = 0; f < 2; ++f)
#pragma unroll
            for (int j = 0; j < 16; ++j) {
                int kk = (j < 8) ? (hi * 8 + j) : (8 + hi * 8 + j);
                a[f][j] = (f16)As[cur][wave * 32 + f * 16 + rr][kk];
            }
#pragma unroll
        for (int t = 0; t < 4; ++t) {
            // B fragment: col N = lane&15; lanes 0-15 K={0..7,16..23}, 16-31 K={8..15,24..31}
            v16h b;
#pragma unroll
            for (int j = 0; j < 16; ++j)
                b[j] = Bs[cur][t * 16 + rr][(j < 8) ? (hi * 8 + j) : (8 + hi * 8 + j)];
#pragma unroll
            for (int f = 0; f < 2; ++f)
                acc[f][t] = __builtin_amdgcn_wmma_f32_16x16x32_f16(
                    false, a[f], false, b, (short)0, acc[f][t], false, false);
        }
        __syncthreads();   // all waves done reading before buffer is overwritten
    }
    // C/D layout: VGPR i -> row i (+8 for lanes 16-31), col = lane&15
#pragma unroll
    for (int f = 0; f < 2; ++f)
#pragma unroll
        for (int t = 0; t < 4; ++t)
#pragma unroll
            for (int i = 0; i < 8; ++i) {
                int row = mBase + wave * 32 + f * 16 + i + hi * 8;
                int col = nBase + t * 16 + rr;
                float v = acc[f][t][i];
                if (RELU) v = fmaxf(v, 0.f);
                C[(size_t)row * N + col] = v;
            }
}

// ---------------------------------------------------- linear attention (per keypoint)
__global__ __launch_bounds__(256)
void attention_kernel(const float* __restrict__ q, const float* __restrict__ k,
                      const float* __restrict__ v, const float* __restrict__ xm,
                      const float* __restrict__ sm, float* __restrict__ msg) {
    __shared__ float Qp[LW * FD];
    __shared__ float Kp[LW * FD];
    __shared__ float Vv[LW * FD];
    __shared__ float KV[8 * 16 * 16];
    __shared__ float Ksum[FD];
    int n = blockIdx.x;
    size_t base = (size_t)n * LW * FD;
    for (int i = threadIdx.x; i < LW * FD; i += 256) {
        int l = i >> 7;
        float mq = xm[n * LW + l], ms = sm[n * LW + l];
        Qp[i] = elu1(q[base + i]) * mq;
        Kp[i] = elu1(k[base + i]) * ms;
        Vv[i] = v[base + i] * ms;
    }
    __syncthreads();
    for (int i = threadIdx.x; i < 2048; i += 256) {      // 8 heads x 16 x 16
        int h = i >> 8, d = (i >> 4) & 15, e = i & 15;
        float s = 0.f;
        for (int l = 0; l < LW; ++l)
            s += Kp[l * FD + h * 16 + d] * Vv[l * FD + h * 16 + e];
        KV[i] = s * (1.0f / (float)LW);
    }
    if (threadIdx.x < FD) {
        float s = 0.f;
        for (int l = 0; l < LW; ++l) s += Kp[l * FD + threadIdx.x];
        Ksum[threadIdx.x] = s;
    }
    __syncthreads();
    for (int i = threadIdx.x; i < LW * FD; i += 256) {
        int l = i >> 7, c = i & 127, h = c >> 4, e = c & 15;
        float zden = ATT_EPS, num = 0.f;
#pragma unroll
        for (int d = 0; d < 16; ++d) {
            float qv = Qp[l * FD + h * 16 + d];
            zden += qv * Ksum[h * 16 + d];
            num += qv * KV[(h * 16 + d) * 16 + e];
        }
        msg[base + i] = num / zden * (float)LW;
    }
}

// ------------------------------------------------- layernorm(msg) -> cat, copy x -> cat
__global__ __launch_bounds__(128)
void ln_cat_kernel(const float* __restrict__ t, const float* __restrict__ x,
                   const float* __restrict__ g, const float* __restrict__ b,
                   float* __restrict__ cat) {
    __shared__ float red[128];
    int row = blockIdx.x, tid = threadIdx.x;
    float v = t[(size_t)row * FD + tid];
    red[tid] = v; __syncthreads();
    for (int s = 64; s > 0; s >>= 1) { if (tid < s) red[tid] += red[tid + s]; __syncthreads(); }
    float mean = red[0] * (1.f / FD);
    __syncthreads();
    float dv = v - mean;
    red[tid] = dv * dv; __syncthreads();
    for (int s = 64; s > 0; s >>= 1) { if (tid < s) red[tid] += red[tid + s]; __syncthreads(); }
    float var = red[0] * (1.f / FD);
    float y = dv * rsqrtf(var + 1e-5f) * g[tid] + b[tid];
    cat[(size_t)row * (2 * FD) + FD + tid] = y;
    cat[(size_t)row * (2 * FD) + tid] = x[(size_t)row * FD + tid];
}

// ------------------------------------------------- layernorm(ffn_out) + residual into x
__global__ __launch_bounds__(128)
void ln_res_kernel(const float* __restrict__ t, const float* __restrict__ g,
                   const float* __restrict__ b, float* __restrict__ x) {
    __shared__ float red[128];
    int row = blockIdx.x, tid = threadIdx.x;
    float v = t[(size_t)row * FD + tid];
    red[tid] = v; __syncthreads();
    for (int s = 64; s > 0; s >>= 1) { if (tid < s) red[tid] += red[tid + s]; __syncthreads(); }
    float mean = red[0] * (1.f / FD);
    __syncthreads();
    float dv = v - mean;
    red[tid] = dv * dv; __syncthreads();
    for (int s = 64; s > 0; s >>= 1) { if (tid < s) red[tid] += red[tid + s]; __syncthreads(); }
    float var = red[0] * (1.f / FD);
    float y = dv * rsqrtf(var + 1e-5f) * g[tid] + b[tid];
    size_t idx = (size_t)row * FD + tid;
    x[idx] = x[idx] + y;
}

// ------------------------------------------------- dual-softmax confidence per keypoint
__global__ __launch_bounds__(256)
void conf_kernel(const float* __restrict__ f0, const float* __restrict__ f1,
                 const float* __restrict__ m0, const float* __restrict__ m1,
                 const float* __restrict__ bin, float* __restrict__ out) {
    __shared__ float F0[LW * FD];
    __shared__ float F1[LW * FD];
    __shared__ float S[26 * 26];
    __shared__ float cmax[26], csum[26], rmax[26], rsum[26];
    int n = blockIdx.x, tid = threadIdx.x;
    size_t base = (size_t)n * LW * FD;
    for (int i = tid; i < LW * FD; i += 256) { F0[i] = f0[base + i]; F1[i] = f1[base + i]; }
    __syncthreads();
    for (int i = tid; i < LW * LW; i += 256) {
        int l = i / LW, s = i % LW;
        float d = 0.f;
#pragma unroll 8
        for (int c = 0; c < FD; ++c) d += F0[l * FD + c] * F1[s * FD + c];
        S[l * 26 + s] = d * (1.0f / 16.0f);   // / sqrt(256) / temp
    }
    float bs = bin[0];
    if (tid < 26) { S[tid * 26 + 25] = bs; S[25 * 26 + tid] = bs; }
    __syncthreads();
    if (tid < 26) {
        float mc = -1e30f, mr = -1e30f;
        for (int l = 0; l < 26; ++l) mc = fmaxf(mc, S[l * 26 + tid]);
        for (int s = 0; s < 26; ++s) mr = fmaxf(mr, S[tid * 26 + s]);
        float sc = 0.f, sr = 0.f;
        for (int l = 0; l < 26; ++l) sc += __expf(S[l * 26 + tid] - mc);
        for (int s = 0; s < 26; ++s) sr += __expf(S[tid * 26 + s] - mr);
        cmax[tid] = mc; csum[tid] = sc; rmax[tid] = mr; rsum[tid] = sr;
    }
    __syncthreads();
    for (int i = tid; i < 26 * 26; i += 256) {
        int l = i / 26, s = i % 26;
        float v = S[i];
        float c = (__expf(v - cmax[s]) / csum[s]) * (__expf(v - rmax[l]) / rsum[l]);
        if (l < LW && s < LW) c *= m0[n * LW + l] * m1[n * LW + s];
        c = fminf(fmaxf(c, 1e-6f), 1.f - 1e-6f);
        out[(size_t)n * 676 + i] = c;
    }
}

// ---------------------------------------------------------------- launcher
extern "C" void kernel_launch(void* const* d_in, const int* in_sizes, int n_in,
                              void* d_out, int out_size, void* d_ws, size_t ws_size,
                              hipStream_t stream) {
    (void)in_sizes; (void)n_in; (void)out_size; (void)ws_size;
    const float* up0  = (const float*)d_in[0];
    const float* up1  = (const float*)d_in[1];
    const float* kp0  = (const float*)d_in[4];
    const float* kp1  = (const float*)d_in[5];
    const float* G1   = (const float*)d_in[12];
    const float* B1   = (const float*)d_in[13];
    const float* G2   = (const float*)d_in[14];
    const float* B2   = (const float*)d_in[15];
    const float* bin  = (const float*)d_in[16];

    float* conf_out = (float*)d_out;
    float* wins0 = conf_out + (size_t)NKP * 26 * 26;
    float* wins1 = wins0 + (size_t)NTOK * 2;

    // ---- workspace carve-up (floats), with aliasing of dead buffers
    float* base = (float*)d_ws;
    size_t off = 0;
    auto FA = [&](size_t nf) { float* p = base + off; off += nf; return p; };
    float* m0   = FA(NTOK);
    float* m1   = FA(NTOK);
    float* f0   = FA((size_t)NTOK * FD);
    float* f1   = FA((size_t)NTOK * FD);
    float* qb   = FA((size_t)NTOK * FD);
    float* kb   = FA((size_t)NTOK * FD);
    float* vb   = FA((size_t)NTOK * FD);
    float* msgb = FA((size_t)NTOK * FD);
    float* tmpb = FA((size_t)NTOK * FD);
    float* hid  = qb;   // alias: q,k dead when FFN hidden is produced
    float* cat  = vb;   // alias: v,msg dead when cat is built
    f16* h16 = (f16*)(base + off);
    size_t ho = 0;
    auto HA = [&](size_t nh) { f16* p = h16 + ho; ho += nh; return p; };
    f16* wq16  = HA((size_t)2 * FD * FD);
    f16* wk16  = HA((size_t)2 * FD * FD);
    f16* wv16  = HA((size_t)2 * FD * FD);
    f16* wm16  = HA((size_t)2 * FD * FD);
    f16* wf116 = HA((size_t)2 * 2 * FD * 2 * FD);
    f16* wf216 = HA((size_t)2 * FD * 2 * FD);

    // ---- weights -> f16 (once per call; cheap)
    auto cvt = [&](const void* s, f16* d, int n) {
        cvt_f16_kernel<<<(n + 255) / 256, 256, 0, stream>>>((const float*)s, d, n);
    };
    cvt(d_in[6],  wq16,  2 * FD * FD);
    cvt(d_in[7],  wk16,  2 * FD * FD);
    cvt(d_in[8],  wv16,  2 * FD * FD);
    cvt(d_in[9],  wm16,  2 * FD * FD);
    cvt(d_in[10], wf116, 2 * 2 * FD * 2 * FD);
    cvt(d_in[11], wf216, 2 * FD * 2 * FD);

    // ---- windows + bilinear sampling -> f0/f1, masks, wins outputs
    sample_kernel<<<NTOK, 128, 0, stream>>>(kp0, up0, f0, m0, wins0);
    sample_kernel<<<NTOK, 128, 0, stream>>>(kp1, up1, f1, m1, wins1);

    // ---- encoder layer (WMMA GEMMs + attention + LN)
    dim3 g128(FD / 64, NTOK / 128);        // N=128 outputs
    dim3 g256(2 * FD / 64, NTOK / 128);    // N=256 outputs
    auto enc = [&](float* x, const float* src, const float* mx, const float* ms, int li) {
        const f16* wq = wq16 + (size_t)li * FD * FD;
        const f16* wk = wk16 + (size_t)li * FD * FD;
        const f16* wv = wv16 + (size_t)li * FD * FD;
        const f16* wm = wm16 + (size_t)li * FD * FD;
        const f16* w1 = wf116 + (size_t)li * 2 * FD * 2 * FD;
        const f16* w2 = wf216 + (size_t)li * FD * 2 * FD;
        gemm_wmma_kernel<128, false><<<g128, 128, 0, stream>>>(x,   wq, qb, FD);
        gemm_wmma_kernel<128, false><<<g128, 128, 0, stream>>>(src, wk, kb, FD);
        gemm_wmma_kernel<128, false><<<g128, 128, 0, stream>>>(src, wv, vb, FD);
        attention_kernel<<<NKP, 256, 0, stream>>>(qb, kb, vb, mx, ms, msgb);
        gemm_wmma_kernel<128, false><<<g128, 128, 0, stream>>>(msgb, wm, tmpb, FD);
        ln_cat_kernel<<<NTOK, 128, 0, stream>>>(tmpb, x, G1 + li * FD, B1 + li * FD, cat);
        gemm_wmma_kernel<256, true ><<<g256, 128, 0, stream>>>(cat, w1, hid, 2 * FD);
        gemm_wmma_kernel<256, false><<<g128, 128, 0, stream>>>(hid, w2, tmpb, FD);
        ln_res_kernel<<<NTOK, 128, 0, stream>>>(tmpb, G2 + li * FD, B2 + li * FD, x);
    };
    enc(f0, f0, m0, m0, 0);   // layer 0: self
    enc(f1, f1, m1, m1, 0);
    enc(f0, f1, m0, m1, 1);   // layer 1: cross (f1 uses updated f0, matching reference)
    enc(f1, f0, m1, m0, 1);

    // ---- dual-softmax confidence
    conf_kernel<<<NKP, 256, 0, stream>>>(f0, f1, m0, m1, bin, conf_out);
}
